// GINE_38311108280997
// MI455X (gfx1250) — matrix operand order repaired and checked by reading, not compiled, not run on hardware
//
#include <hip/hip_runtime.h>

#define HID 256
#define ATOM_FDIM 133
#define BOND_RAW 15
#define MAXNB 16
#define BN_EPS 1e-5f

typedef __attribute__((ext_vector_type(16))) __bf16 bf16x16;
typedef __attribute__((ext_vector_type(8)))  __bf16 bf16x8;
typedef __attribute__((ext_vector_type(8)))  float  f32x8;

__device__ __forceinline__ f32x8 wmma_bf16f32(bf16x16 a, bf16x16 b, f32x8 c) {
  return __builtin_amdgcn_wmma_f32_16x16x32_bf16(false, a, false, b, (short)0, c, false, false);
}

__device__ __forceinline__ bf16x16 cat8(bf16x8 lo, bf16x8 hi) {
  bf16x16 a;
#pragma unroll
  for (int i = 0; i < 8; ++i) { a[i] = lo[i]; a[8 + i] = hi[i]; }
  return a;
}

// ---- fragment loaders (all-bf16, branch-free, 16B-aligned vector loads) -----
// A: bf16 [*, ldk] row-major, K pre-padded to multiple of 32. Row clamped to
// Mm1 (garbage rows only feed masked-off outputs). ISA layout: lane<16 ->
// m=lane, K={kb..kb+7, kb+16..kb+23}, kb=k0; lane>=16 -> kb=k0+8.
__device__ __forceinline__ bf16x16 load_a_frag(const __bf16* A, int ldk, int m0,
                                               int Mm1, int k0) {
  int lane = threadIdx.x & 31;
  int m = m0 + (lane & 15);
  m = m > Mm1 ? Mm1 : m;
  const __bf16* p = A + (long long)m * ldk + k0 + ((lane >> 4) << 3);
  return cat8(*(const bf16x8*)p, *(const bf16x8*)(p + 16));
}

// A from per-wave bf16 LDS tile T[16][ldk]
__device__ __forceinline__ bf16x16 load_a_frag_lds(const __bf16* T, int ldk, int k0) {
  int lane = threadIdx.x & 31;
  const __bf16* p = T + (lane & 15) * ldk + k0 + ((lane >> 4) << 3);
  return cat8(*(const bf16x8*)p, *(const bf16x8*)(p + 16));
}

// B: Wt bf16 [Nn][ldk] (pre-transposed, K zero-padded). lane<16: n=n0+lane,
// K={kb..kb+7, kb+16..kb+23}, kb=k0; lane>=16: kb=k0+8.
__device__ __forceinline__ bf16x16 load_b_frag(const __bf16* Wt, int ldk, int n0, int k0) {
  int lane = threadIdx.x & 31;
  const __bf16* p = Wt + (long long)(n0 + (lane & 15)) * ldk + k0 + ((lane >> 4) << 3);
  return cat8(*(const bf16x8*)p, *(const bf16x8*)(p + 16));
}

// ---- plain GEMM: C[M,Nn] = A[M,Kpad]@W + bias. One wave = 16x64 tile. -------
__global__ void gemm_kernel(const __bf16* __restrict__ A, int Kpad,
                            const __bf16* __restrict__ Wt, const float* __restrict__ bias,
                            int Nn, float* __restrict__ C, int M, int do_relu) {
  int wid  = (int)((blockIdx.x * blockDim.x + threadIdx.x) >> 5);
  int lane = threadIdx.x & 31;
  int nblk = Nn >> 6;
  int mt = wid / nblk, nb = wid % nblk;
  int m0 = mt * 16;
  if (m0 >= M) return;
  f32x8 acc0 = {}, acc1 = {}, acc2 = {}, acc3 = {};
  for (int k0 = 0; k0 < Kpad; k0 += 32) {
    bf16x16 a = load_a_frag(A, Kpad, m0, M - 1, k0);
    acc0 = wmma_bf16f32(a, load_b_frag(Wt, Kpad, nb * 64 +  0, k0), acc0);
    acc1 = wmma_bf16f32(a, load_b_frag(Wt, Kpad, nb * 64 + 16, k0), acc1);
    acc2 = wmma_bf16f32(a, load_b_frag(Wt, Kpad, nb * 64 + 32, k0), acc2);
    acc3 = wmma_bf16f32(a, load_b_frag(Wt, Kpad, nb * 64 + 48, k0), acc3);
  }
  int mb = m0 + ((lane >> 4) << 3);
  f32x8 accs[4] = {acc0, acc1, acc2, acc3};
#pragma unroll
  for (int j = 0; j < 4; ++j) {
    int n = nb * 64 + j * 16 + (lane & 15);
    float bv = bias ? bias[n] : 0.0f;
#pragma unroll
    for (int r = 0; r < 8; ++r) {
      int m = mb + r;
      if (m < M) {
        float v = accs[j][r] + bv;
        if (do_relu) v = fmaxf(v, 0.0f);
        C[(long long)m * Nn + n] = v;
      }
    }
  }
}

// ---- fused 2-layer MLP: out = (relu(A@W1+b1))@W2+b2, hidden staged in LDS ----
// mode 0: store Out[M,N2]. mode 1 (bond): out += Hg[b2a[m]]; atomicAdd into
// Aggr[to[m]] (fused message + segment_sum).
__global__ void fused_mlp_kernel(const __bf16* __restrict__ A, int K0pad,
                                 const __bf16* __restrict__ W1t, const float* __restrict__ b1, int H1,
                                 const __bf16* __restrict__ W2t, const float* __restrict__ b2, int N2,
                                 int M, int mode,
                                 float* __restrict__ Out,
                                 const float* __restrict__ Hg,
                                 const int* __restrict__ b2a,
                                 const int* __restrict__ to,
                                 float* __restrict__ Aggr) {
  extern __shared__ __bf16 smem[];
  int wave = threadIdx.x >> 5;
  int lane = threadIdx.x & 31;
  __bf16* T = smem + (size_t)wave * 16 * H1;
  int mt = blockIdx.x * (blockDim.x >> 5) + wave;
  int m0 = mt * 16;
  if (m0 >= M) return;  // uniform per-wave exit; EXEC stays full for WMMA
  int mb = (lane >> 4) << 3;

  // stage 1: T[16][H1] = relu(A_tile @ W1 + b1), bf16 in per-wave LDS slice
  for (int nt = 0; nt < (H1 >> 4); ++nt) {
    f32x8 c = {};
    for (int k0 = 0; k0 < K0pad; k0 += 32) {
      bf16x16 a = load_a_frag(A, K0pad, m0, M - 1, k0);
      bf16x16 b = load_b_frag(W1t, K0pad, nt << 4, k0);
      c = wmma_bf16f32(a, b, c);
    }
    int n = (nt << 4) + (lane & 15);
    float bv = b1[n];
#pragma unroll
    for (int r = 0; r < 8; ++r)
      T[(mb + r) * H1 + n] = (__bf16)fmaxf(c[r] + bv, 0.0f);
  }

  // hoist per-row indices for the scatter epilogue
  int srcs[8], dsts[8];
  if (mode == 1) {
#pragma unroll
    for (int r = 0; r < 8; ++r) {
      int m = m0 + mb + r;
      if (m > M - 1) m = M - 1;
      srcs[r] = b2a[m];
      dsts[r] = to[m];
    }
  }

  // stage 2 (wave-private LDS tile; DS waits handled by compiler)
  for (int nt = 0; nt < (N2 >> 4); ++nt) {
    f32x8 c = {};
    for (int k0 = 0; k0 < H1; k0 += 32) {
      bf16x16 a = load_a_frag_lds(T, H1, k0);
      bf16x16 b = load_b_frag(W2t, H1, nt << 4, k0);
      c = wmma_bf16f32(a, b, c);
    }
    int n = (nt << 4) + (lane & 15);
    float bv = b2[n];
    if (mode == 0) {
#pragma unroll
      for (int r = 0; r < 8; ++r) {
        int m = m0 + mb + r;
        if (m < M) Out[(long long)m * N2 + n] = c[r] + bv;
      }
    } else {
#pragma unroll
      for (int r = 0; r < 8; ++r) {
        int m = m0 + mb + r;
        if (m < M) {
          float v = c[r] + bv + Hg[(long long)srcs[r] * HID + n];
          atomicAdd(&Aggr[(long long)dsts[r] * HID + n], v);
        }
      }
    }
  }
}

// ---- small helper kernels ---------------------------------------------------
__global__ void zero_i32_kernel(int* p, long long n) {
  long long i = (long long)blockIdx.x * blockDim.x + threadIdx.x;
  if (i < n) p[i] = 0;
}
__global__ void zero_f32_kernel(float* p, long long n) {
  long long i = (long long)blockIdx.x * blockDim.x + threadIdx.x;
  if (i < n) p[i] = 0.0f;
}
__global__ void build_to_kernel(const int* __restrict__ a2b, int* __restrict__ to, long long n) {
  long long i = (long long)blockIdx.x * blockDim.x + threadIdx.x;
  if (i < n) {
    int b = a2b[i];
    if (b > 0) to[b] = (int)(i >> 4);  // atom index = i / MAXNB
  }
}
// W[K][Nn] fp32 -> Wt[Nn][Kpad] bf16 (K zero-padded)
__global__ void convert_wt_kernel(const float* __restrict__ W, int K, int Nn, int Kpad,
                                  __bf16* __restrict__ Wt) {
  long long i = (long long)blockIdx.x * blockDim.x + threadIdx.x;
  if (i < (long long)Nn * Kpad) {
    int n = (int)(i / Kpad), k = (int)(i % Kpad);
    Wt[i] = (k < K) ? (__bf16)W[(long long)k * Nn + n] : (__bf16)0.0f;
  }
}
// A[M,K] fp32 -> B[M,Kpad] bf16 (K zero-padded)
__global__ void convert_act_kernel(const float* __restrict__ A, int K, int Kpad,
                                   __bf16* __restrict__ B, long long total) {
  long long i = (long long)blockIdx.x * blockDim.x + threadIdx.x;
  if (i < total) {
    int k = (int)(i % Kpad);
    long long m = i / Kpad;
    B[i] = (k < K) ? (__bf16)A[m * K + k] : (__bf16)0.0f;
  }
}
// self-loop embedding: sl = relu(w1[15,:] + b1) @ w2 + b2   (one block, 256 thr)
__global__ void sl_emb_kernel(const float* __restrict__ w1, const float* __restrict__ b1,
                              const float* __restrict__ w2, const float* __restrict__ b2,
                              float* __restrict__ sl) {
  __shared__ float t[HID];
  int c = threadIdx.x;
  t[c] = fmaxf(w1[15 * HID + c] + b1[c], 0.0f);
  __syncthreads();
  float acc = b2[c];
  for (int k = 0; k < HID; ++k) acc += t[k] * w2[k * HID + c];
  sl[c] = acc;
}
__global__ void aggr_init_kernel(const float* __restrict__ h, const float* __restrict__ sl,
                                 float* __restrict__ aggr, long long n) {
  long long i = (long long)blockIdx.x * blockDim.x + threadIdx.x;
  if (i < n) aggr[i] = h[i] + sl[i & (HID - 1)];
}
// column sums / sums of squares over M rows into sums[0..255] / sums[256..511]
__global__ void bn_stats_kernel(const float* __restrict__ z, int M, float* __restrict__ sums) {
  int c = threadIdx.x;
  int per = (M + gridDim.x - 1) / gridDim.x;
  long long r0 = (long long)blockIdx.x * per;
  long long r1 = r0 + per; if (r1 > M) r1 = M;
  float s = 0.0f, ss = 0.0f;
  for (long long r = r0; r < r1; ++r) {
    float v = z[r * HID + c];
    s += v; ss += v * v;
  }
  atomicAdd(&sums[c], s);
  atomicAdd(&sums[HID + c], ss);
}
__global__ void bn_apply_kernel(const float* __restrict__ z, const float* __restrict__ sums,
                                const float* __restrict__ gamma, const float* __restrict__ beta,
                                float* __restrict__ out, int M, int do_relu) {
  long long i = (long long)blockIdx.x * blockDim.x + threadIdx.x;
  long long n = (long long)M * HID;
  if (i >= n) return;
  int c = (int)(i & (HID - 1));
  float inv = 1.0f / (float)M;
  float mu  = sums[c] * inv;
  float var = sums[HID + c] * inv - mu * mu;
  float v = gamma[c] * (z[i] - mu) * rsqrtf(var + BN_EPS) + beta[c];
  if (do_relu) v = fmaxf(v, 0.0f);
  out[i] = v;
}

// ---- driver -----------------------------------------------------------------
extern "C" void kernel_launch(void* const* d_in, const int* in_sizes, int n_in,
                              void* d_out, int out_size, void* d_ws, size_t ws_size,
                              hipStream_t stream) {
  const float* f_atoms = (const float*)d_in[0];
  const float* f_bonds = (const float*)d_in[1];
  const int*   a2b     = (const int*)d_in[2];
  const int*   b2a     = (const int*)d_in[3];
  const float* w_atom  = (const float*)d_in[6];
  const float* wb_w1   = (const float*)d_in[7];
  const float* wb_b1   = (const float*)d_in[8];
  const float* wb_w2   = (const float*)d_in[9];
  const float* wb_b2   = (const float*)d_in[10];
  const float* m_w1    = (const float*)d_in[11];
  const float* m_b1    = (const float*)d_in[12];
  const float* m_w2    = (const float*)d_in[13];
  const float* m_b2    = (const float*)d_in[14];
  const float* bn_g    = (const float*)d_in[15];
  const float* bn_b    = (const float*)d_in[16];
  const int N = in_sizes[0] / ATOM_FDIM;
  const int E = in_sizes[1] / BOND_RAW;

  char* ws = (char*)d_ws;
  size_t off = 0;
  auto alloc = [&](size_t bytes) -> char* {
    char* p = ws + off;
    off = (off + bytes + 255) & ~(size_t)255;
    return p;
  };
  int*     to      = (int*)    alloc((size_t)E * 4);
  __bf16*  wAtomT  = (__bf16*) alloc((size_t)256 * 160 * 2);
  __bf16*  w1T     = (__bf16*) alloc((size_t)3 * 256 * 32  * 2);
  __bf16*  w2T     = (__bf16*) alloc((size_t)3 * 256 * 256 * 2);
  __bf16*  mw1T    = (__bf16*) alloc((size_t)3 * 512 * 256 * 2);
  __bf16*  mw2T    = (__bf16*) alloc((size_t)3 * 256 * 512 * 2);
  float*   sl      = (float*)  alloc(256 * 4);
  float*   bnsums  = (float*)  alloc(512 * 4);
  float*   h       = (float*)  alloc((size_t)N * HID * 4);
  float*   aggr    = (float*)  alloc((size_t)N * HID * 4);
  float*   zb      = (float*)  alloc((size_t)N * HID * 4);
  __bf16*  fAbf    = (__bf16*) alloc((size_t)N * 160 * 2);   // f_atoms padded bf16
  __bf16*  fBbf    = (__bf16*) alloc((size_t)E * 32 * 2);    // f_bonds padded bf16
  __bf16*  aggrbf  = (__bf16*) alloc((size_t)N * HID * 2);   // aggr bf16
  (void)n_in; (void)out_size; (void)ws_size;

  // edge target reconstruction
  {
    long long n = E;
    zero_i32_kernel<<<(unsigned)((n + 255) / 256), 256, 0, stream>>>(to, n);
  }
  {
    long long n = (long long)N * MAXNB;
    build_to_kernel<<<(unsigned)((n + 255) / 256), 256, 0, stream>>>(a2b, to, n);
  }
  // weight conversion fp32 -> transposed, K-padded bf16
  auto conv = [&](const float* W, int K, int Nn, int Kpad, __bf16* Wt) {
    long long n = (long long)Nn * Kpad;
    convert_wt_kernel<<<(unsigned)((n + 255) / 256), 256, 0, stream>>>(W, K, Nn, Kpad, Wt);
  };
  conv(w_atom, ATOM_FDIM, 256, 160, wAtomT);
  for (int l = 0; l < 3; ++l) {
    conv(wb_w1 + (size_t)l * 16 * 256,  16,  256, 32,  w1T  + (size_t)l * 256 * 32);
    conv(wb_w2 + (size_t)l * 256 * 256, 256, 256, 256, w2T  + (size_t)l * 256 * 256);
    conv(m_w1  + (size_t)l * 256 * 512, 256, 512, 256, mw1T + (size_t)l * 512 * 256);
    conv(m_w2  + (size_t)l * 512 * 256, 512, 256, 512, mw2T + (size_t)l * 256 * 512);
  }
  // activation conversions (once)
  {
    long long n = (long long)N * 160;
    convert_act_kernel<<<(unsigned)((n + 255) / 256), 256, 0, stream>>>(f_atoms, ATOM_FDIM, 160, fAbf, n);
  }
  {
    long long n = (long long)E * 32;
    convert_act_kernel<<<(unsigned)((n + 255) / 256), 256, 0, stream>>>(f_bonds, BOND_RAW, 32, fBbf, n);
  }
  // atom embedding: h = f_atoms @ w_atom
  {
    int Mtiles = (N + 15) / 16;
    long long waves = (long long)Mtiles * (256 / 64);
    unsigned blocks = (unsigned)((waves + 7) / 8);
    gemm_kernel<<<blocks, 256, 0, stream>>>(fAbf, 160, wAtomT, nullptr, 256, h, N, 0);
  }
  for (int l = 0; l < 3; ++l) {
    sl_emb_kernel<<<1, 256, 0, stream>>>(wb_w1 + (size_t)l * 16 * 256,
                                         wb_b1 + (size_t)l * 256,
                                         wb_w2 + (size_t)l * 256 * 256,
                                         wb_b2 + (size_t)l * 256, sl);
    {
      long long n = (long long)N * HID;
      aggr_init_kernel<<<(unsigned)((n + 255) / 256), 256, 0, stream>>>(h, sl, aggr, n);
    }
    // bond MLP + gather + scatter-add (mode 1)
    {
      int Mtiles = (E + 15) / 16;
      int wpb = 8;
      unsigned blocks = (unsigned)((Mtiles + wpb - 1) / wpb);
      size_t shmem = (size_t)wpb * 16 * 256 * 2;  // 64 KB
      fused_mlp_kernel<<<blocks, wpb * 32, shmem, stream>>>(
          fBbf, 32,
          w1T + (size_t)l * 256 * 32,  wb_b1 + (size_t)l * 256, 256,
          w2T + (size_t)l * 256 * 256, wb_b2 + (size_t)l * 256, 256,
          E, 1, nullptr, h, b2a, to, aggr);
    }
    // aggr -> bf16 for node MLP
    {
      long long n = (long long)N * HID;
      convert_act_kernel<<<(unsigned)((n + 255) / 256), 256, 0, stream>>>(aggr, HID, HID, aggrbf, n);
    }
    // node MLP -> zb (mode 0)
    {
      int Mtiles = (N + 15) / 16;
      int wpb = 4;
      unsigned blocks = (unsigned)((Mtiles + wpb - 1) / wpb);
      size_t shmem = (size_t)wpb * 16 * 512 * 2;  // 64 KB
      fused_mlp_kernel<<<blocks, wpb * 32, shmem, stream>>>(
          aggrbf, 256,
          mw1T + (size_t)l * 512 * 256, m_b1 + (size_t)l * 512, 512,
          mw2T + (size_t)l * 256 * 512, m_b2 + (size_t)l * 256, 256,
          N, 0, zb, nullptr, nullptr, nullptr, nullptr);
    }
    // BatchNorm (training-mode batch stats)
    zero_f32_kernel<<<2, 256, 0, stream>>>(bnsums, 512);
    bn_stats_kernel<<<512, 256, 0, stream>>>(zb, N, bnsums);
    float* outp = (l < 2) ? h : (float*)d_out;
    {
      long long n = (long long)N * HID;
      bn_apply_kernel<<<(unsigned)((n + 255) / 256), 256, 0, stream>>>(
          zb, bnsums, bn_g + (size_t)l * 256, bn_b + (size_t)l * 256, outp, N, (l < 2) ? 1 : 0);
    }
  }
}
